// InterFrameAttention_83631603188468
// MI455X (gfx1250) — matrix operand order repaired and checked by reading, not compiled, not run on hardware
//
#include <hip/hip_runtime.h>

typedef __attribute__((ext_vector_type(16))) _Float16 v16h;
typedef __attribute__((ext_vector_type(8)))  _Float16 v8h;
typedef __attribute__((ext_vector_type(4)))  _Float16 v4h;
typedef __attribute__((ext_vector_type(8)))  float    v8f;

// ---------------------------------------------------------------------------
// WMMA wrapper: D = A(16x32 f16) * B(32x16 f16) + C(16x16 f32)
// ---------------------------------------------------------------------------
__device__ __forceinline__ v8f wmma32(v16h a, v16h b, v8f c) {
  return __builtin_amdgcn_wmma_f32_16x16x32_f16(
      /*neg_a=*/false, a, /*neg_b=*/false, b,
      /*c_mod=*/(short)0, c, /*reuse_a=*/false, /*reuse_b=*/false);
}

// ---------------------------------------------------------------------------
// Fragment loaders (wave32); this code's contiguous K order, same for A and B (not the ISA table order).
//  A 16x32 f16: lane holds row r=lane&15; half hh=lane>>4 holds
//    K in {16*hh..16*hh+15}  (two 16B runs)
//  B 32x16 f16: lane holds col n=lane&15; half hh holds K in {16*hh..16*hh+15}
// ---------------------------------------------------------------------------
__device__ __forceinline__ v16h frag_a(const _Float16* rowk, int hh) {
  const v8h lo = *(const v8h*)(rowk + 16 * hh);
  const v8h hi = *(const v8h*)(rowk + 16 * hh + 8);
  v16h x;
#pragma unroll
  for (int j = 0; j < 8; ++j) { x[j] = lo[j]; x[8 + j] = hi[j]; }
  return x;
}

__device__ __forceinline__ v16h frag_bt(const _Float16* colk, int hh) {
  const v8h lo = *(const v8h*)(colk + 16 * hh);
  const v8h hi = *(const v8h*)(colk + 16 * hh + 8);
  v16h x;
#pragma unroll
  for (int j = 0; j < 8; ++j) { x[j] = lo[j]; x[8 + j] = hi[j]; }
  return x;
}

__device__ __forceinline__ v16h frag_b_strided(const _Float16* col, int ldb, int hh) {
  v16h x;
  const _Float16* p = col + (size_t)(16 * hh) * ldb;
#pragma unroll
  for (int j = 0; j < 16; ++j) x[j] = p[(size_t)j * ldb];
  return x;
}

// C/D 16x16 f32: VGPR v holds row (v + 8*hh), col = lane&15
__device__ __forceinline__ void store_f16(_Float16* D, int ld, int mBase, int nBase,
                                          int r, int hh, v8f c) {
#pragma unroll
  for (int v = 0; v < 8; ++v)
    D[(size_t)(mBase + v + 8 * hh) * ld + nBase + r] = (_Float16)c[v];
}

// ---------------------------------------------------------------------------
// Prep kernel: convert + transpose weights to f16 in workspace.
//   wts[0      ..65536 )  qWt   [256n][256k]
//   wts[65536  ..196608)  kvWt  [512n][256k]
//   wts[196608 ..262144)  projWt[256n][256k]
//   wts[262144 ..278528)  motWt [128n][128k]
// ---------------------------------------------------------------------------
__global__ void ifa_prep(const float* __restrict__ q_w, const float* __restrict__ kv_w,
                         const float* __restrict__ proj_w, const float* __restrict__ motion_w,
                         _Float16* __restrict__ wts) {
  int stride = gridDim.x * blockDim.x;
  for (int i = blockIdx.x * blockDim.x + threadIdx.x; i < 278528; i += stride) {
    if (i < 65536) {
      int n = i >> 8, k = i & 255;
      wts[i] = (_Float16)q_w[k * 256 + n];
    } else if (i < 196608) {
      int j = i - 65536; int n = j >> 8, k = j & 255;
      wts[i] = (_Float16)kv_w[k * 512 + n];
    } else if (i < 262144) {
      int j = i - 196608; int n = j >> 8, k = j & 255;
      wts[i] = (_Float16)proj_w[k * 256 + n];
    } else {
      int j = i - 262144; int n = j >> 7, k = j & 127;
      wts[i] = (_Float16)motion_w[k * 128 + n];
    }
  }
}

// ---------------------------------------------------------------------------
// Main kernel: 1 block = 1 batch window; 256 threads = 8 waves = 8 heads.
// Dynamic LDS (192KB):
//   [0      ..65536 ) kvh   f16 [64][512]
//   [65536  ..98304 ) qaf   f16 [64][256]   (q, later reused per-wave as af)
//   [98304  ..131072) x1h   f16 [64][256]   } reused as
//   [131072 ..163840) x2h   f16 [64][256]   } attnh f16 [8][64][64]
//   [163840 ..180224) ceh   f16 [64][128]
//   [180224 ..196608) mfh   f16 [64][128]
// ---------------------------------------------------------------------------
__global__ void __launch_bounds__(256, 1)
ifa_main(const float* __restrict__ x1, const float* __restrict__ x2,
         const float* __restrict__ cor, const float* __restrict__ mask,
         const float* __restrict__ q_bias, const float* __restrict__ kv_bias,
         const float* __restrict__ cor_w, const float* __restrict__ cor_b,
         const float* __restrict__ proj_bias, const float* __restrict__ motion_bias,
         const _Float16* __restrict__ wts, float* __restrict__ out,
         int nBatch, int nMaskW) {
  extern __shared__ char smem[];
  _Float16* kvh   = (_Float16*)(smem);
  _Float16* qaf   = (_Float16*)(smem + 65536);
  _Float16* x1h   = (_Float16*)(smem + 98304);
  _Float16* x2h   = (_Float16*)(smem + 131072);
  _Float16* attnh = (_Float16*)(smem + 98304);
  _Float16* ceh   = (_Float16*)(smem + 163840);
  _Float16* mfh   = (_Float16*)(smem + 180224);

  const int tid  = threadIdx.x;
  const int wave = tid >> 5;          // == head
  const int lane = tid & 31;
  const int r    = lane & 15;
  const int hh   = lane >> 4;
  const int blk  = blockIdx.x;

  const _Float16* qWt    = wts;
  const _Float16* kvWt   = wts + 65536;
  const _Float16* projWt = wts + 196608;
  const _Float16* motWt  = wts + 262144;

  // ---- Phase 1: stage x1/x2 as f16; compute cor_e (K=2) as f16 ------------
  {
    const float4* x1g = (const float4*)(x1 + (size_t)blk * 16384);
    const float4* x2g = (const float4*)(x2 + (size_t)blk * 16384);
    for (int i = tid; i < 4096; i += 256) {
      float4 a = x1g[i];
      v4h pa = { (_Float16)a.x, (_Float16)a.y, (_Float16)a.z, (_Float16)a.w };
      *(v4h*)(x1h + i * 4) = pa;
      float4 b = x2g[i];
      v4h pb = { (_Float16)b.x, (_Float16)b.y, (_Float16)b.z, (_Float16)b.w };
      *(v4h*)(x2h + i * 4) = pb;
    }
    const float* corB = cor + (size_t)blk * 128;
    for (int i = tid; i < 8192; i += 256) {
      int row = i >> 7, m = i & 127;
      float v = corB[row * 2] * cor_w[m] + corB[row * 2 + 1] * cor_w[128 + m] + cor_b[m];
      ceh[i] = (_Float16)v;
    }
  }
  __syncthreads();

  const float scale = 0.17677669529663687f;   // 1/sqrt(32), folded into q

  // ---- Phase 2: q = (x1 @ q_w + q_b) * scale  -> qaf f16 -------------------
  // 2 N-tiles per A-frag: 2 independent WMMA chains, half the A ds-loads.
  {
    const int nt0 = wave * 2, nt1 = nt0 + 1;
    const float bias0 = q_bias[nt0 * 16 + r];
    const float bias1 = q_bias[nt1 * 16 + r];
#pragma unroll
    for (int mt = 0; mt < 4; ++mt) {
      v8f c0 = {}, c1 = {};
#pragma unroll
      for (int kt = 0; kt < 8; ++kt) {
        v16h a  = frag_a(x1h + (size_t)(mt * 16 + r) * 256 + kt * 32, hh);
        v16h b0 = frag_bt(qWt + (size_t)(nt0 * 16 + r) * 256 + kt * 32, hh);
        v16h b1 = frag_bt(qWt + (size_t)(nt1 * 16 + r) * 256 + kt * 32, hh);
        c0 = wmma32(a, b0, c0);
        c1 = wmma32(a, b1, c1);
      }
#pragma unroll
      for (int v = 0; v < 8; ++v) {
        c0[v] = (c0[v] + bias0) * scale;
        c1[v] = (c1[v] + bias1) * scale;
      }
      store_f16(qaf, 256, mt * 16, nt0 * 16, r, hh, c0);
      store_f16(qaf, 256, mt * 16, nt1 * 16, r, hh, c1);
    }
  }

  // ---- Phase 3: kv = x2 @ kv_w + kv_b  -> kvh f16 --------------------------
  // 4 N-tiles per A-frag: 4 independent WMMA chains.
  {
    const int ntb = wave * 4;
    float bias[4];
#pragma unroll
    for (int j = 0; j < 4; ++j) bias[j] = kv_bias[(ntb + j) * 16 + r];
#pragma unroll
    for (int mt = 0; mt < 4; ++mt) {
      v8f c[4] = {{}, {}, {}, {}};
#pragma unroll
      for (int kt = 0; kt < 8; ++kt) {
        v16h a = frag_a(x2h + (size_t)(mt * 16 + r) * 256 + kt * 32, hh);
#pragma unroll
        for (int j = 0; j < 4; ++j) {
          v16h b = frag_bt(kvWt + (size_t)((ntb + j) * 16 + r) * 256 + kt * 32, hh);
          c[j] = wmma32(a, b, c[j]);
        }
      }
#pragma unroll
      for (int j = 0; j < 4; ++j) {
#pragma unroll
        for (int v = 0; v < 8; ++v) c[j][v] += bias[j];
        store_f16(kvh, 512, mt * 16, (ntb + j) * 16, r, hh, c[j]);
      }
    }
  }
  __syncthreads();

  // ---- Phase 4: scores = q_h @ k_h^T (+mask), register softmax -> attnh f16
  const int head = wave;
  _Float16* attnH = attnh + head * 4096;
  {
    const float* maskW = mask + (size_t)(blk % nMaskW) * 4096;
#pragma unroll
    for (int mt = 0; mt < 4; ++mt) {
      v16h a = frag_a(qaf + (size_t)(mt * 16 + r) * 256 + head * 32, hh);
      v8f c[4];
#pragma unroll
      for (int nt = 0; nt < 4; ++nt) {
        // B[k][n] = k_h[n][k] = kvh[n][head*32+k]: 16 contiguous K per column
        v16h b = frag_bt(kvh + (size_t)(nt * 16 + r) * 512 + head * 32, hh);
        v8f z = {};
        c[nt] = wmma32(a, b, z);
      }
#pragma unroll
      for (int v = 0; v < 8; ++v) {
        const int row = mt * 16 + v + 8 * hh;
        const float* mrow = maskW + row * 64;
        float t0 = c[0][v] + mrow[r];
        float t1 = c[1][v] + mrow[16 + r];
        float t2 = c[2][v] + mrow[32 + r];
        float t3 = c[3][v] + mrow[48 + r];
        float pm = fmaxf(fmaxf(t0, t1), fmaxf(t2, t3));
        pm = fmaxf(pm, __shfl_xor(pm, 1));
        pm = fmaxf(pm, __shfl_xor(pm, 2));
        pm = fmaxf(pm, __shfl_xor(pm, 4));
        pm = fmaxf(pm, __shfl_xor(pm, 8));
        t0 = __expf(t0 - pm); t1 = __expf(t1 - pm);
        t2 = __expf(t2 - pm); t3 = __expf(t3 - pm);
        float s = t0 + t1 + t2 + t3;
        s += __shfl_xor(s, 1);
        s += __shfl_xor(s, 2);
        s += __shfl_xor(s, 4);
        s += __shfl_xor(s, 8);
        const float inv = 1.0f / s;
        _Float16* arow = attnH + row * 64;
        arow[r]      = (_Float16)(t0 * inv);
        arow[16 + r] = (_Float16)(t1 * inv);
        arow[32 + r] = (_Float16)(t2 * inv);
        arow[48 + r] = (_Float16)(t3 * inv);
      }
    }
  }

  // ---- Phase 5 (+5b merged): af_h = attn @ v_h ; mf_h = attn @ ce_h - ce_h
  // One attn A-frag feeds 3 independent WMMA chains.
  {
#pragma unroll
    for (int mt = 0; mt < 4; ++mt) {
      v8f c0 = {}, c1 = {}, c2 = {};
#pragma unroll
      for (int kt = 0; kt < 2; ++kt) {
        v16h a  = frag_a(attnH + (size_t)(mt * 16 + r) * 64 + kt * 32, hh);
        v16h b0 = frag_b_strided(
            kvh + (size_t)(kt * 32) * 512 + 256 + head * 32 + r, 512, hh);
        v16h b1 = frag_b_strided(
            kvh + (size_t)(kt * 32) * 512 + 256 + head * 32 + 16 + r, 512, hh);
        v16h b2 = frag_b_strided(ceh + (size_t)(kt * 32) * 128 + head * 16 + r, 128, hh);
        c0 = wmma32(a, b0, c0);
        c1 = wmma32(a, b1, c1);
        c2 = wmma32(a, b2, c2);
      }
      store_f16(qaf, 256, mt * 16, head * 32, r, hh, c0);
      store_f16(qaf, 256, mt * 16, head * 32 + 16, r, hh, c1);
#pragma unroll
      for (int v = 0; v < 8; ++v) {
        const int row = mt * 16 + v + 8 * hh;
        c2[v] -= (float)ceh[(size_t)row * 128 + head * 16 + r];
      }
      store_f16(mfh, 128, mt * 16, head * 16, r, hh, c2);
    }
  }
  __syncthreads();

  // ---- Phase 6: af_out = af @ proj_w + proj_b  -> global f32 --------------
  float* outAf = out + (size_t)blk * 16384;
  {
    const int nt0 = wave * 2, nt1 = nt0 + 1;
    const float bias0 = proj_bias[nt0 * 16 + r];
    const float bias1 = proj_bias[nt1 * 16 + r];
#pragma unroll
    for (int mt = 0; mt < 4; ++mt) {
      v8f c0 = {}, c1 = {};
#pragma unroll
      for (int kt = 0; kt < 8; ++kt) {
        v16h a  = frag_a(qaf + (size_t)(mt * 16 + r) * 256 + kt * 32, hh);
        v16h b0 = frag_bt(projWt + (size_t)(nt0 * 16 + r) * 256 + kt * 32, hh);
        v16h b1 = frag_bt(projWt + (size_t)(nt1 * 16 + r) * 256 + kt * 32, hh);
        c0 = wmma32(a, b0, c0);
        c1 = wmma32(a, b1, c1);
      }
#pragma unroll
      for (int v = 0; v < 8; ++v) {
        const size_t row = (size_t)(mt * 16 + v + 8 * hh) * 256;
        outAf[row + nt0 * 16 + r] = c0[v] + bias0;
        outAf[row + nt1 * 16 + r] = c1[v] + bias1;
      }
    }
  }

  // ---- Phase 7: mf_out = mf @ motion_w + motion_b  -> global f32 ----------
  // Share the B-frag across 4 M-tiles: 4 independent WMMA chains.
  float* outMf = out + (size_t)nBatch * 16384 + (size_t)blk * 8192;
  {
    const int nt = wave;                       // 8 waves cover N=128
    const float bias = motion_bias[nt * 16 + r];
    v8f c[4] = {{}, {}, {}, {}};
#pragma unroll
    for (int kt = 0; kt < 4; ++kt) {
      v16h b = frag_bt(motWt + (size_t)(nt * 16 + r) * 128 + kt * 32, hh);
#pragma unroll
      for (int mt = 0; mt < 4; ++mt) {
        v16h a = frag_a(mfh + (size_t)(mt * 16 + r) * 128 + kt * 32, hh);
        c[mt] = wmma32(a, b, c[mt]);
      }
    }
#pragma unroll
    for (int mt = 0; mt < 4; ++mt) {
#pragma unroll
      for (int v = 0; v < 8; ++v)
        outMf[(size_t)(mt * 16 + v + 8 * hh) * 128 + nt * 16 + r] = c[mt][v] + bias;
    }
  }
}

// ---------------------------------------------------------------------------
extern "C" void kernel_launch(void* const* d_in, const int* in_sizes, int n_in,
                              void* d_out, int out_size, void* d_ws, size_t ws_size,
                              hipStream_t stream) {
  const float* x1       = (const float*)d_in[0];
  const float* x2       = (const float*)d_in[1];
  const float* cor      = (const float*)d_in[2];
  const float* mask     = (const float*)d_in[3];
  const float* q_w      = (const float*)d_in[4];
  const float* q_b      = (const float*)d_in[5];
  const float* kv_w     = (const float*)d_in[6];
  const float* kv_b     = (const float*)d_in[7];
  const float* cor_w    = (const float*)d_in[8];
  const float* cor_b    = (const float*)d_in[9];
  const float* proj_w   = (const float*)d_in[10];
  const float* proj_b   = (const float*)d_in[11];
  const float* motion_w = (const float*)d_in[12];
  const float* motion_b = (const float*)d_in[13];

  const int nBatch = in_sizes[0] / (64 * 256);     // 2048
  const int nMaskW = in_sizes[3] / (64 * 64);      // 512
  _Float16* wts = (_Float16*)d_ws;                 // 278528 f16 = 544KB

  ifa_prep<<<272, 256, 0, stream>>>(q_w, kv_w, proj_w, motion_w, wts);

  const size_t shmem = 196608;                     // 192KB dynamic LDS
  (void)hipFuncSetAttribute((const void*)ifa_main,
                            hipFuncAttributeMaxDynamicSharedMemorySize, (int)shmem);
  ifa_main<<<nBatch, 256, shmem, stream>>>(x1, x2, cor, mask, q_b, kv_b, cor_w, cor_b,
                                           proj_b, motion_b, wts, (float*)d_out,
                                           nBatch, nMaskW);
}
